// DenoiseNetwork_16638703304704
// MI455X (gfx1250) — matrix operand order repaired and checked by reading, not compiled, not run on hardware
//
#include <hip/hip_runtime.h>
#include <math.h>

// ---------------- problem constants (from reference) ----------------
#define BB      4
#define NNODES  128          // context nodes
#define NM      32           // target nodes
#define TT      512
#define CC      64
#define LL      3
#define KTAP    3
#define NGRP    4
#define CT      (CC*TT)      // 32768 flattened (c,t) per node

typedef __attribute__((ext_vector_type(16))) _Float16 v16h;
typedef __attribute__((ext_vector_type(8)))  float    v8f;

// A-fragment K index (contiguous order, matching the B fills):
// lanes 0-15: K=0..15 ; lanes 16-31: K=16..31
__device__ __forceinline__ int akidx(int j, int hl) { return j + 16*hl; }

__device__ __forceinline__ v8f wmma_f16(v16h a, v16h b, v8f c) {
  return __builtin_amdgcn_wmma_f32_16x16x32_f16(false, a, false, b, (short)0, c,
                                                false, false);
}

// ---------------- elementwise: feature embedding (dy == 1) ----------------
// out[b,n,c,t] = Wfe[c]*y[b,n,0,t] + bfe[c]
__global__ void feat_embed_kernel(const float* __restrict__ y,
                                  const float* __restrict__ Wfe,
                                  const float* __restrict__ bfe,
                                  int total, float* __restrict__ out) {
  int idx = blockIdx.x * blockDim.x + threadIdx.x;
  if (idx >= total) return;
  int t  = idx & (TT - 1);
  int c  = (idx >> 9) & (CC - 1);
  int bn = idx >> 15;
  out[idx] = Wfe[c] * y[bn * TT + t] + bfe[c];
}

// ---------------- diffusion embedding (tiny MLP), all layers ----------------
__global__ void diff_emb_kernel(const int* __restrict__ dstep,
                                const float* __restrict__ Wp1, const float* __restrict__ bp1,
                                const float* __restrict__ Wp2, const float* __restrict__ bp2,
                                float* __restrict__ emb) {
  int li = blockIdx.x / BB;
  int b  = blockIdx.x % BB;
  int c  = threadIdx.x;                       // 0..63
  __shared__ float e0[CC], h1s[CC];
  float s = (float)dstep[b];
  int j = (c < 32) ? c : (c - 32);
  float fr = powf(10.f, (float)j * 4.f / 31.f);
  e0[c] = (c < 32) ? sinf(s * fr) : cosf(s * fr);
  __syncthreads();
  const float* W1 = Wp1 + li * CC * CC;
  float acc = bp1[li * CC + c];
  for (int k = 0; k < CC; ++k) acc += W1[c * CC + k] * e0[k];
  acc = acc / (1.f + expf(-acc));             // SiLU
  h1s[c] = acc;
  __syncthreads();
  const float* W2 = Wp2 + li * CC * CC;
  float a2 = bp2[li * CC + c];
  for (int k = 0; k < CC; ++k) a2 += W2[c * CC + k] * h1s[k];
  a2 = a2 / (1.f + expf(-a2));
  emb[(li * BB + b) * CC + c] = a2;
}

// Te[b,m,c,t] = Tin + emb[b,c]
__global__ void add_emb_kernel(const float* __restrict__ x, const float* __restrict__ embL,
                               float* __restrict__ out, int total) {
  int idx = blockIdx.x * blockDim.x + threadIdx.x;
  if (idx >= total) return;
  int c = (idx >> 9) & (CC - 1);
  int b = idx >> 20;                           // NM*CC*TT == 2^20
  out[idx] = x[idx] + embL[b * CC + c];
}

// ---------------- WMMA: forward hop  h[b,m,ct] = adj^T (32x128) @ src(*mask) ----------------
// grid (CT/16, BB), block 32. One wave owns all M=32 rows (2 tiles): the expensive
// strided B gather is built ONCE per K-step and reused for both A tiles.
template <bool UseMask>
__global__ void hop_fwd_kernel(const float* __restrict__ adj, const float* __restrict__ src,
                               const float* __restrict__ mask, float* __restrict__ out) {
  int lane = threadIdx.x & 31, hl = lane >> 4, r = lane & 15;
  int col = blockIdx.x * 16 + r;               // flattened (c,t)
  int b   = blockIdx.y;
  int t   = col & (TT - 1);
  const float* adjb = adj + (size_t)b * NNODES * NM;
  const float* srcb = src + (size_t)b * NNODES * CT;
  const float* mb   = mask + (size_t)b * NNODES * TT;   // unused when !UseMask
  v8f acc[2] = {{}, {}};
#pragma unroll
  for (int k0 = 0; k0 < NNODES; k0 += 32) {    // K = 128 context nodes
    v16h bx;
#pragma unroll
    for (int j = 0; j < 16; ++j) {
      int nb = k0 + hl * 16 + j;
      float x = srcb[(size_t)nb * CT + col];
      if (UseMask) x *= mb[nb * TT + t];
      bx[j] = (_Float16)x;
    }
#pragma unroll
    for (int mt = 0; mt < 2; ++mt) {           // M tiles share the B fragment
      v16h a;
#pragma unroll
      for (int j = 0; j < 16; ++j)
        a[j] = (_Float16)adjb[(k0 + akidx(j, hl)) * NM + mt * 16 + r]; // A[m][n]=adj[n][m]
      acc[mt] = wmma_f16(a, bx, acc[mt]);
    }
  }
  float* ob = out + (size_t)b * NM * CT;
#pragma unroll
  for (int mt = 0; mt < 2; ++mt)
#pragma unroll
    for (int v = 0; v < 8; ++v)
      ob[(size_t)(mt * 16 + v + 8 * hl) * CT + col] = acc[mt][v];
}

// ---------------- WMMA: backward hop  back[b,n,ct] = (adj(128x32) @ h1) * mask ----------------
// grid (CT/16, BB), block 32. One wave owns all M=128 rows (8 tiles, K=32 -> 8 WMMA);
// the h1 column fragment is gathered once and reused 8x.
__global__ void hop_back_kernel(const float* __restrict__ adj, const float* __restrict__ h1,
                                const float* __restrict__ mask, float* __restrict__ out) {
  int lane = threadIdx.x & 31, hl = lane >> 4, r = lane & 15;
  int col = blockIdx.x * 16 + r;
  int b   = blockIdx.y;
  int t   = col & (TT - 1);
  const float* adjb = adj + (size_t)b * NNODES * NM;
  const float* h1b  = h1  + (size_t)b * NM * CT;
  const float* mb   = mask + (size_t)b * NNODES * TT;
  v16h bx;
#pragma unroll
  for (int j = 0; j < 16; ++j)
    bx[j] = (_Float16)h1b[(size_t)(hl * 16 + j) * CT + col];
  float* ob = out + (size_t)b * NNODES * CT;
#pragma unroll
  for (int nt = 0; nt < 8; ++nt) {
    int n0 = nt * 16;
    v16h a;
#pragma unroll
    for (int j = 0; j < 16; ++j)
      a[j] = (_Float16)adjb[(n0 + r) * NM + akidx(j, hl)];
    v8f acc = {};
    acc = wmma_f16(a, bx, acc);
#pragma unroll
    for (int v = 0; v < 8; ++v) {
      int n = n0 + v + 8 * hl;
      ob[(size_t)n * CT + col] = acc[v] * mb[n * TT + t];
    }
  }
}

// ---------------- WMMA: graph-agg mix  G = Wga(64x192) @ [Te;h1;h2] + bga ----------------
// grid (TT/16, BB*NM), block 32. One wave owns all O=64 rows (4 tiles): activation
// fragment built once per K-step, reused by 4 weight tiles. 24 WMMA / wave.
__global__ void ga_mix_kernel(const float* __restrict__ Wga_, const float* __restrict__ bga_,
                              const float* __restrict__ te, const float* __restrict__ h1,
                              const float* __restrict__ h2, float* __restrict__ out) {
  int lane = threadIdx.x & 31, hl = lane >> 4, r = lane & 15;
  int t0 = blockIdx.x * 16;
  size_t base = (size_t)blockIdx.y * CT;
  const float* S0 = te + base;
  const float* S1 = h1 + base;
  const float* S2 = h2 + base;
  v8f acc[4] = {{}, {}, {}, {}};
#pragma unroll
  for (int k0 = 0; k0 < 3 * CC; k0 += 32) {    // K = 192
    const float* S = (k0 < CC) ? S0 : ((k0 < 2 * CC) ? S1 : S2);
    int cb0 = k0 & (CC - 1);
    v16h bx;
#pragma unroll
    for (int j = 0; j < 16; ++j)
      bx[j] = (_Float16)S[(cb0 + hl * 16 + j) * TT + t0 + r];
#pragma unroll
    for (int ot = 0; ot < 4; ++ot) {
      v16h a;
#pragma unroll
      for (int j = 0; j < 16; ++j)
        a[j] = (_Float16)Wga_[(ot * 16 + r) * (3 * CC) + k0 + akidx(j, hl)];
      acc[ot] = wmma_f16(a, bx, acc[ot]);
    }
  }
  float* Ob = out + base;
#pragma unroll
  for (int ot = 0; ot < 4; ++ot)
#pragma unroll
    for (int v = 0; v < 8; ++v) {
      int o = ot * 16 + v + 8 * hl;
      Ob[(size_t)o * TT + t0 + r] = acc[ot][v] + bga_[o];
    }
}

// ---------------- WMMA: fused causal gated dilated conv (+ residual 1x1) ----------------
// out = tanh(Wf*X+bf) * sigmoid(Wg*X+bg) [+ Wres*Res + bres]
// grid (TT/16, BB*nodes), block 32. One wave owns all O=64 rows: each shifted X
// fragment is gathered once and feeds 8 WMMAs (4 o-tiles x {filter,gate}).
template <bool HasRes>
__global__ void gated_tconv_kernel(const float* __restrict__ X,   const float* __restrict__ Res,
                                   const float* __restrict__ Wf_, const float* __restrict__ bf_,
                                   const float* __restrict__ Wg_, const float* __restrict__ bg_,
                                   const float* __restrict__ Wr_, const float* __restrict__ br_,
                                   int dil, float* __restrict__ out) {
  int lane = threadIdx.x & 31, hl = lane >> 4, r = lane & 15;
  int t0 = blockIdx.x * 16;
  const float* Xb = X + (size_t)blockIdx.y * CT;
  int tcol = t0 + r;
  v8f accF[4] = {{}, {}, {}, {}};
  v8f accG[4] = {{}, {}, {}, {}};
  v8f accR[4] = {{}, {}, {}, {}};
#pragma unroll
  for (int k = 0; k < KTAP; ++k) {
    int ts  = tcol - (KTAP - 1 - k) * dil;     // causal shift
    bool ok = ts >= 0;
    int tsc = ok ? ts : 0;                     // clamp: never form OOB address
#pragma unroll
    for (int c0 = 0; c0 < CC; c0 += 32) {
      v16h bx;
#pragma unroll
      for (int j = 0; j < 16; ++j) {
        float xv = Xb[(c0 + hl * 16 + j) * TT + tsc];
        bx[j] = ok ? (_Float16)xv : (_Float16)0.f;
      }
#pragma unroll
      for (int ot = 0; ot < 4; ++ot) {
        v16h af, ag;
#pragma unroll
        for (int j = 0; j < 16; ++j) {
          int widx = ((ot * 16 + r) * CC + c0 + akidx(j, hl)) * KTAP + k;
          af[j] = (_Float16)Wf_[widx];
          ag[j] = (_Float16)Wg_[widx];
        }
        accF[ot] = wmma_f16(af, bx, accF[ot]);
        accG[ot] = wmma_f16(ag, bx, accG[ot]);
      }
    }
  }
  if (HasRes) {
    const float* Rb = Res + (size_t)blockIdx.y * CT;
#pragma unroll
    for (int c0 = 0; c0 < CC; c0 += 32) {
      v16h bx;
#pragma unroll
      for (int j = 0; j < 16; ++j)
        bx[j] = (_Float16)Rb[(c0 + hl * 16 + j) * TT + tcol];
#pragma unroll
      for (int ot = 0; ot < 4; ++ot) {
        v16h ar;
#pragma unroll
        for (int j = 0; j < 16; ++j)
          ar[j] = (_Float16)Wr_[(ot * 16 + r) * CC + c0 + akidx(j, hl)];
        accR[ot] = wmma_f16(ar, bx, accR[ot]);
      }
    }
  }
  float* Ob = out + (size_t)blockIdx.y * CT;
#pragma unroll
  for (int ot = 0; ot < 4; ++ot)
#pragma unroll
    for (int v = 0; v < 8; ++v) {
      int o = ot * 16 + v + 8 * hl;
      float f = accF[ot][v] + bf_[o];
      float g = accG[ot][v] + bg_[o];
      float y = tanhf(f) * (1.f / (1.f + expf(-g)));
      if (HasRes) y += accR[ot][v] + br_[o];
      Ob[(size_t)o * TT + tcol] = y;
    }
}

// ---------------- GroupNorm: stats (atomic-reduced) + in-place normalize ----------------
__global__ void zero_kernel(float* p, int n) {
  int i = blockIdx.x * blockDim.x + threadIdx.x;
  if (i < n) p[i] = 0.f;
}

// sums[bg*2] += sum, sums[bg*2+1] += sumsq over (c in group, n, t). cpg*TT == 8192 == 2^13.
__global__ void gn_stats_kernel(const float* __restrict__ X, int nn, float* __restrict__ sums) {
  int bg = blockIdx.x;
  int b = bg / NGRP, g = bg % NGRP;
  const int cpg = CC / NGRP;                   // 16
  size_t tot = (size_t)cpg * nn * TT;
  size_t per = (tot + gridDim.y - 1) / gridDim.y;
  size_t s0 = (size_t)blockIdx.y * per;
  size_t s1 = s0 + per; if (s1 > tot) s1 = tot;
  float s = 0.f, sq = 0.f;
  for (size_t idx = s0 + threadIdx.x; idx < s1; idx += blockDim.x) {
    int t  = (int)(idx & (TT - 1));
    int ci = (int)((idx >> 9) & (cpg - 1));
    int n  = (int)(idx >> 13);
    float x = X[(((size_t)b * nn + n) * CC + g * cpg + ci) * TT + t];
    s += x; sq += x * x;
  }
  __shared__ float sh[256], shq[256];
  int tid = threadIdx.x;
  sh[tid] = s; shq[tid] = sq;
  __syncthreads();
  for (int off = 128; off > 0; off >>= 1) {
    if (tid < off) { sh[tid] += sh[tid + off]; shq[tid] += shq[tid + off]; }
    __syncthreads();
  }
  if (tid == 0) {
    atomicAdd(&sums[bg * 2 + 0], sh[0]);
    atomicAdd(&sums[bg * 2 + 1], shq[0]);
  }
}

__global__ void gn_norm_kernel(float* __restrict__ X, const float* __restrict__ sums,
                               const float* __restrict__ gammaL, const float* __restrict__ betaL,
                               int nn, int total) {
  int idx = blockIdx.x * blockDim.x + threadIdx.x;
  if (idx >= total) return;
  int c = (idx >> 9) & (CC - 1);
  int b = idx / (nn * CT);
  int g = c / (CC / NGRP);
  int bg = b * NGRP + g;
  float cnt = (float)((CC / NGRP) * nn * TT);
  float mu  = sums[bg * 2 + 0] / cnt;
  float var = sums[bg * 2 + 1] / cnt - mu * mu;
  float inv = rsqrtf(var + 1e-5f);
  X[idx] = (X[idx] - mu) * inv * gammaL[c] + betaL[c];
}

// ---------------------------------- driver ----------------------------------
extern "C" void kernel_launch(void* const* d_in, const int* in_sizes, int n_in,
                              void* d_out, int out_size, void* d_ws, size_t ws_size,
                              hipStream_t stream) {
  (void)in_sizes; (void)n_in; (void)out_size; (void)ws_size;
  const float* y_context = (const float*)d_in[1];
  const float* y_target  = (const float*)d_in[3];
  const float* adj       = (const float*)d_in[4];
  const float* mask      = (const float*)d_in[5];
  const int*   step      = (const int*)  d_in[6];
  const float* W_fe  = (const float*)d_in[7];
  const float* b_fe  = (const float*)d_in[8];
  const float* Wp1   = (const float*)d_in[9];
  const float* bp1   = (const float*)d_in[10];
  const float* Wp2   = (const float*)d_in[11];
  const float* bp2   = (const float*)d_in[12];
  const float* Wga   = (const float*)d_in[13];
  const float* bga   = (const float*)d_in[14];
  const float* Wf    = (const float*)d_in[15];
  const float* bf    = (const float*)d_in[16];
  const float* Wg    = (const float*)d_in[17];
  const float* bg    = (const float*)d_in[18];
  const float* Wres  = (const float*)d_in[19];
  const float* bres  = (const float*)d_in[20];
  const float* gamma = (const float*)d_in[21];
  const float* beta  = (const float*)d_in[22];

  const size_t CTX = (size_t)BB * NNODES * CT;   // 16,777,216 floats per layer (context)
  const size_t TGT = (size_t)BB * NM * CT;       //  4,194,304 floats per layer (target)

  float* dc = (float*)d_out;                     // stacked d_c : [L][b][n][c][t]
  float* dt = dc + LL * CTX;                     // stacked d_t : [L][b][m][c][t]

  // Scratch aliased onto not-yet-written output slots (write-before-read, per call):
  float* ctx0 = dc + 1 * CTX;  // initial context; overwritten by layer-1 context output
  float* tgt0 = dt + 1 * TGT;  // initial target;  overwritten by layer-1 target  output
  float* back = dc + 2 * CTX;  // hop-back buffer; overwritten by layer-2 context output
  // True workspace (~67 MB):
  float* Te    = (float*)d_ws;            // target + emb (residual source)
  float* h1    = Te + TGT;
  float* h2    = h1 + TGT;
  float* Gb    = h2 + TGT;                // graph-agg mixed output
  float* emb   = Gb + TGT;                // [L][B][C]
  float* stats = emb + LL * BB * CC;      // [16 pairs target][16 pairs context]

  feat_embed_kernel<<<(int)(CTX / 256), 256, 0, stream>>>(y_context, W_fe, b_fe, (int)CTX, ctx0);
  feat_embed_kernel<<<(int)(TGT / 256), 256, 0, stream>>>(y_target,  W_fe, b_fe, (int)TGT, tgt0);
  diff_emb_kernel<<<LL * BB, CC, 0, stream>>>(step, Wp1, bp1, Wp2, bp2, emb);

  for (int i = 0; i < LL; ++i) {
    const float* Tin = (i == 0) ? tgt0 : dt + (size_t)(i - 1) * TGT;
    const float* Cin = (i == 0) ? ctx0 : dc + (size_t)(i - 1) * CTX;
    const float* Wf_l = Wf   + (size_t)i * CC * CC * KTAP;
    const float* Wg_l = Wg   + (size_t)i * CC * CC * KTAP;
    const float* Wr_l = Wres + (size_t)i * CC * CC;
    const float* bf_l = bf + i * CC, *bg_l = bg + i * CC, *br_l = bres + i * CC;
    float* outT = dt + (size_t)i * TGT;
    float* outC = dc + (size_t)i * CTX;

    add_emb_kernel<<<(int)(TGT / 256), 256, 0, stream>>>(Tin, emb + i * BB * CC, Te, (int)TGT);
    hop_fwd_kernel<true> <<<dim3(CT / 16, BB), 32, 0, stream>>>(adj, Cin,  mask, h1);
    hop_back_kernel      <<<dim3(CT / 16, BB), 32, 0, stream>>>(adj, h1,   mask, back);
    hop_fwd_kernel<false><<<dim3(CT / 16, BB), 32, 0, stream>>>(adj, back, mask, h2);
    ga_mix_kernel<<<dim3(TT / 16, BB * NM), 32, 0, stream>>>(
        Wga + (size_t)i * CC * 3 * CC, bga + i * CC, Te, h1, h2, Gb);

    if (i == 0) {
      gated_tconv_kernel<false><<<dim3(TT / 16, BB * NM), 32, 0, stream>>>(
          Gb, nullptr, Wf_l, bf_l, Wg_l, bg_l, Wr_l, br_l, 1, outT);
      gated_tconv_kernel<false><<<dim3(TT / 16, BB * NNODES), 32, 0, stream>>>(
          Cin, nullptr, Wf_l, bf_l, Wg_l, bg_l, Wr_l, br_l, 1, outC);
    } else {
      gated_tconv_kernel<true><<<dim3(TT / 16, BB * NM), 32, 0, stream>>>(
          Gb, Te, Wf_l, bf_l, Wg_l, bg_l, Wr_l, br_l, 1 << i, outT);
      gated_tconv_kernel<true><<<dim3(TT / 16, BB * NNODES), 32, 0, stream>>>(
          Cin, Cin, Wf_l, bf_l, Wg_l, bg_l, Wr_l, br_l, 1 << i, outC);

      zero_kernel<<<1, 64, 0, stream>>>(stats, 64);
      gn_stats_kernel<<<dim3(BB * NGRP, 32),  256, 0, stream>>>(outT, NM,     stats);
      gn_stats_kernel<<<dim3(BB * NGRP, 128), 256, 0, stream>>>(outC, NNODES, stats + 32);
      gn_norm_kernel<<<(int)(TGT / 256), 256, 0, stream>>>(outT, stats,      gamma + i * CC, beta + i * CC, NM,     (int)TGT);
      gn_norm_kernel<<<(int)(CTX / 256), 256, 0, stream>>>(outC, stats + 32, gamma + i * CC, beta + i * CC, NNODES, (int)CTX);
    }
  }
}
